// LightweightLinearAttention2d_2336462209422
// MI455X (gfx1250) — matrix operand order repaired and checked by reading, not compiled, not run on hardware
//
#include <hip/hip_runtime.h>

typedef __attribute__((ext_vector_type(16))) _Float16 v16h;
typedef __attribute__((ext_vector_type(8)))  _Float16 v8h;
typedef __attribute__((ext_vector_type(8)))  float    v8f;
typedef __attribute__((ext_vector_type(4)))  int      v4i;
typedef __attribute__((ext_vector_type(8)))  int      v8i;
typedef unsigned int u32;
typedef unsigned long long u64;

// ---------------------------------------------------------------------------
// WMMA helpers (CDNA5 gfx1250, wave32). D = A(16x32 f16) * B(32x16 f16) + C(f32)
// ---------------------------------------------------------------------------
__device__ __forceinline__ v8f wmma_f16(v16h a, v16h b, v8f c) {
  return __builtin_amdgcn_wmma_f32_16x16x32_f16(
      /*neg_a=*/false, a, /*neg_b=*/false, b,
      /*c_mod=*/(short)0, c, /*reuse_a=*/false, /*reuse_b=*/false);
}

// A fragment: 16(M) x 32(K) from row-major f16 source, leading dim ld (elems).
// ISA layout: lanes 0-15 row=lane, K = {0..7,16..23}; lanes 16-31 row=lane-16,
// K = {8..15,24..31}. Two contiguous 16-byte vector loads per lane.
__device__ __forceinline__ v16h load_a_frag(const _Float16* tile, int ld) {
  int lane = threadIdx.x & 31;
  int row  = lane & 15;
  int kb   = (lane & 16) ? 8 : 0;
  const _Float16* p = tile + (size_t)row * ld + kb;
  v8h lo = *(const v8h*)p;
  v8h hi = *(const v8h*)(p + 16);
  return __builtin_shufflevector(lo, hi, 0,1,2,3,4,5,6,7,8,9,10,11,12,13,14,15);
}

// B fragment: 32(K) x 16(N), from row-major (K x N) source, leading dim ld.
// Lane l<16: col=l, K={0..7,16..23} ; lane l>=16: col=l-16, K={8..15,24..31}.
__device__ __forceinline__ v16h load_b_frag(const _Float16* tile, int ld) {
  int lane = threadIdx.x & 31;
  int col  = lane & 15;
  int kb   = (lane & 16) ? 8 : 0;
  v16h b;
  #pragma unroll
  for (int i = 0; i < 16; ++i) b[i] = tile[(size_t)(i < 8 ? kb + i : kb + 8 + i) * ld + col];
  return b;
}

// ---------------------------------------------------------------------------
// Tensor Data Mover: 2D tile load (global -> LDS), D# per ISA ch.8.
// group0: count=1, lds_addr, 57-bit global tile addr, type=2 ("image").
// group1: data_size=2B, tensor_dim0/1, tile_dim0/1, dim0 stride.
// groups 2/3: zero-filled (tile_dim3/4 = 0 -> unused).
// Issued by one wave; completion via TENSORcnt.
// ---------------------------------------------------------------------------
__device__ __forceinline__ void tdm_load_2d_f16(u64 gaddr, u32 lds_addr,
                                                u32 tile_w, u32 tile_h,
                                                u32 stride_elems,
                                                u32 tensor_w, u32 tensor_h) {
  v4i g0;
  g0[0] = 1;                                             // count=1, user D#
  g0[1] = (int)lds_addr;                                 // LDS byte address
  g0[2] = (int)(u32)gaddr;                               // global addr [31:0]
  g0[3] = (int)((((u32)(gaddr >> 32)) & 0x01FFFFFFu) | 0x80000000u); // [56:32]|type=2
  v8i g1;
  g1[0] = 0x00010000;                                    // data_size=1 (2 bytes)
  g1[1] = (int)(tensor_w << 16);                         // tensor_dim0[15:0]
  g1[2] = (int)((tensor_w >> 16) | (tensor_h << 16));    // dim0 hi | dim1 lo
  g1[3] = (int)((tensor_h >> 16) | (tile_w << 16));      // dim1 hi | tile_dim0
  g1[4] = (int)(tile_h & 0xFFFFu);                       // tile_dim1 (tile_dim2=0)
  g1[5] = (int)stride_elems;                             // tensor_dim0_stride lo32
  g1[6] = 0;                                             // stride hi | dim1_stride lo
  g1[7] = 0;
  v4i gz = {0, 0, 0, 0};
  asm volatile("tensor_load_to_lds %0, %1, %2, %3"
               :: "s"(g0), "s"(g1), "s"(gz), "s"(gz)
               : "memory");
}

// ---------------------------------------------------------------------------
// Stage 0: converts / init
// ---------------------------------------------------------------------------
__global__ void __launch_bounds__(256) cvt_f32_f16(const float* __restrict__ s,
                                                   _Float16* __restrict__ d, int n) {
  int i = blockIdx.x * 256 + threadIdx.x;
  if (i < n) d[i] = (_Float16)s[i];
}

__global__ void __launch_bounds__(256) zero_f32(float* __restrict__ p, int n) {
  int i = blockIdx.x * 256 + threadIdx.x;
  if (i < n) p[i] = 0.0f;
}

__global__ void __launch_bounds__(256) bn_prep(const float* __restrict__ g,
                                               const float* __restrict__ be,
                                               const float* __restrict__ mu,
                                               const float* __restrict__ var,
                                               float* __restrict__ scale,
                                               float* __restrict__ shift) {
  int i = blockIdx.x * 256 + threadIdx.x;
  if (i < 384) {
    float s = g[i] * rsqrtf(var[i] + 1e-5f);
    scale[i] = s;
    shift[i] = be[i] - mu[i] * s;
  }
}

// Qp/Vp are [b*h][4096][64]: d-columns 48..63 are padding.
// Qp pad = 0 (so K-padding contributes nothing); Vp col 48 = 1.0 (ksum trick),
// cols 49..63 = 0.
__global__ void __launch_bounds__(256) pad_qv(_Float16* __restrict__ Qp,
                                              _Float16* __restrict__ Vp) {
  int i = blockIdx.x * 256 + threadIdx.x;       // 64*4096 rows
  size_t base = (size_t)i * 64;
  #pragma unroll
  for (int j = 0; j < 16; ++j) {
    Qp[base + 48 + j] = (_Float16)0.0f;
    Vp[base + 48 + j] = (_Float16)((j == 0) ? 1.0f : 0.0f);
  }
}

// ---------------------------------------------------------------------------
// Stage 1: qkv = Wqkv(1152x384) @ X(384x4096) per batch  -> f16 [b][1152][4096]
// 8 waves/block = 4x2 grid of 32x32 wave tiles -> 128x64 block tile.
// B tile (32K x 64N) staged to LDS by the Tensor Data Mover each K-chunk;
// A fragments load straight from global (two b128 per lane).
// ---------------------------------------------------------------------------
__global__ void __launch_bounds__(256) gemm_qkv(const _Float16* __restrict__ W,
                                                const _Float16* __restrict__ X,
                                                _Float16* __restrict__ out) {
  __shared__ __align__(16) _Float16 Bt[32 * 64];
  int b  = blockIdx.z;
  int w  = threadIdx.x >> 5;
  int wx = w >> 1, wy = w & 1;
  int m0 = blockIdx.x * 128 + wx * 32;
  int nb = blockIdx.y * 64;                     // block N window
  const _Float16* Bb = X + (size_t)b * 384 * 4096;
  u32 lds = (u32)(size_t)&Bt[0];
  v8f c00 = {}, c01 = {}, c10 = {}, c11 = {};
  for (int kc = 0; kc < 384; kc += 32) {
    __syncthreads();                            // readers of previous tile done
    if (__builtin_amdgcn_readfirstlane(w) == 0) {
      u64 g = (u64)(size_t)(Bb + (size_t)kc * 4096 + nb);
      tdm_load_2d_f16(g, lds, /*tile_w=*/64, /*tile_h=*/32,
                      /*stride=*/4096, /*tensor_w=*/4096, /*tensor_h=*/4096);
      __builtin_amdgcn_s_wait_tensorcnt(0);
    }
    __syncthreads();
    v16h a0 = load_a_frag(W + (size_t)m0 * 384 + kc, 384);
    v16h a1 = load_a_frag(W + (size_t)(m0 + 16) * 384 + kc, 384);
    v16h b0 = load_b_frag(Bt + wy * 32, 64);
    v16h b1 = load_b_frag(Bt + wy * 32 + 16, 64);
    c00 = wmma_f16(a0, b0, c00);
    c01 = wmma_f16(a0, b1, c01);
    c10 = wmma_f16(a1, b0, c10);
    c11 = wmma_f16(a1, b1, c11);
  }
  int lane = threadIdx.x & 31;
  int col = lane & 15, rb = (lane & 16) ? 8 : 0;
  int n0 = nb + wy * 32;
  _Float16* ob = out + (size_t)b * 1152 * 4096;
  #pragma unroll
  for (int r = 0; r < 8; ++r) {
    ob[(size_t)(m0 + rb + r) * 4096 + n0 + col]           = (_Float16)c00[r];
    ob[(size_t)(m0 + rb + r) * 4096 + n0 + 16 + col]      = (_Float16)c01[r];
    ob[(size_t)(m0 + 16 + rb + r) * 4096 + n0 + col]      = (_Float16)c10[r];
    ob[(size_t)(m0 + 16 + rb + r) * 4096 + n0 + 16 + col] = (_Float16)c11[r];
  }
}

// ---------------------------------------------------------------------------
// Stage 2: z = qkv + 0.5*(dw3 + dw5); split/route with layout transforms.
//   q (c<384)    -> relu -> Qp [bh][n][64]   (WMMA-A for out GEMM)
//   k (384..767) -> relu -> Kt [b][384][n]   (WMMA-A for kv GEMM)
//   v (768..)    ->         Vp [bh][n][64]   (WMMA-B for kv GEMM)
// 3x3 weights folded into the 5x5 window (single pass).
// ---------------------------------------------------------------------------
__global__ void __launch_bounds__(256) dwconv_route(const _Float16* __restrict__ qkvh,
                                                    const float* __restrict__ w3,
                                                    const float* __restrict__ w5,
                                                    _Float16* __restrict__ Qp,
                                                    _Float16* __restrict__ Kt,
                                                    _Float16* __restrict__ Vp) {
  int idx = blockIdx.x * 256 + threadIdx.x;     // 8*1152*4096 total
  int n = idx & 4095;
  int t = idx >> 12;
  int c = t % 1152;
  int b = t / 1152;
  int px = n & 63, py = n >> 6;
  const _Float16* base = qkvh + (size_t)(b * 1152 + c) * 4096;
  const float* W3 = w3 + c * 9;
  const float* W5 = w5 + c * 25;
  float acc = 0.0f;
  #pragma unroll
  for (int dy = -2; dy <= 2; ++dy) {
    int yy = py + dy;
    #pragma unroll
    for (int dx = -2; dx <= 2; ++dx) {
      int xx = px + dx;
      float v = 0.0f;
      if (yy >= 0 && yy < 64 && xx >= 0 && xx < 64)
        v = (float)base[yy * 64 + xx];
      float wgt = W5[(dy + 2) * 5 + (dx + 2)];
      if (dy >= -1 && dy <= 1 && dx >= -1 && dx <= 1)
        wgt += W3[(dy + 1) * 3 + (dx + 1)];
      acc += wgt * v;
    }
  }
  float z = (float)base[n] + 0.5f * acc;
  if (c < 384) {
    int h = c / 48, d = c - h * 48;
    Qp[((size_t)(b * 8 + h) * 4096 + n) * 64 + d] = (_Float16)fmaxf(z, 0.0f);
  } else if (c < 768) {
    int cc = c - 384;
    Kt[((size_t)b * 384 + cc) * 4096 + n] = (_Float16)fmaxf(z, 0.0f);
  } else {
    int cc = c - 768;
    int h = cc / 48, d = cc - h * 48;
    Vp[((size_t)(b * 8 + h) * 4096 + n) * 64 + d] = (_Float16)z;
  }
}

// ---------------------------------------------------------------------------
// Stage 3: kv_aug[48][64] = K^T(48x4096) @ Vaug(4096x64) per (b,h).
// Column 48 of Vaug is 1.0 -> kv_aug[:,48] = ksum. 4096-deep reduction split
// over grid.y segments and 4 waves; fp32 global atomic accumulation.
// ---------------------------------------------------------------------------
__global__ void __launch_bounds__(128) gemm_kv(const _Float16* __restrict__ Kt,
                                               const _Float16* __restrict__ Vp,
                                               float* __restrict__ kvacc) {
  int bh = blockIdx.x, seg = blockIdx.y;
  int w = threadIdx.x >> 5;
  int b = bh >> 3, h = bh & 7;
  const _Float16* Ab = Kt + ((size_t)b * 384 + h * 48) * 4096;
  const _Float16* Bb = Vp + (size_t)bh * 4096 * 64;
  v8f acc[3][4] = {};
  for (int j = 0; j < 8; ++j) {
    int nk = (seg * 32 + w * 8 + j) * 32;       // 128 K-chunks of 32
    v16h a[3];
    #pragma unroll
    for (int ti = 0; ti < 3; ++ti)
      a[ti] = load_a_frag(Ab + (size_t)ti * 16 * 4096 + nk, 4096);
    v16h bf[4];
    #pragma unroll
    for (int e = 0; e < 4; ++e)
      bf[e] = load_b_frag(Bb + (size_t)nk * 64 + e * 16, 64);
    #pragma unroll
    for (int ti = 0; ti < 3; ++ti)
      #pragma unroll
      for (int e = 0; e < 4; ++e)
        acc[ti][e] = wmma_f16(a[ti], bf[e], acc[ti][e]);
  }
  int lane = threadIdx.x & 31;
  int col = lane & 15, rb = (lane & 16) ? 8 : 0;
  float* kb = kvacc + (size_t)bh * 48 * 64;
  #pragma unroll
  for (int ti = 0; ti < 3; ++ti)
    #pragma unroll
    for (int e = 0; e < 4; ++e)
      #pragma unroll
      for (int r = 0; r < 8; ++r)
        atomicAdd(&kb[(size_t)(ti * 16 + rb + r) * 64 + e * 16 + col], acc[ti][e][r]);
}

// kvacc fp32 [bh][48][64] -> kvh f16 [bh][64][64], K-rows 48..63 zero-padded.
__global__ void __launch_bounds__(256) cvt_kv(const float* __restrict__ acc,
                                              _Float16* __restrict__ kvh) {
  int i = blockIdx.x * 256 + threadIdx.x;       // 64*64*64
  int bh = i >> 12;
  int r = (i >> 6) & 63;
  int e = i & 63;
  kvh[i] = (r < 48) ? (_Float16)acc[(size_t)bh * 3072 + r * 64 + e] : (_Float16)0.0f;
}

// ---------------------------------------------------------------------------
// Stage 4: Out_aug = Qp(4096x64) @ kv_aug(64x64) per (b,h).
// Column 48 of the result is the normalizer q.ksum; broadcast via __shfl,
// divide, pack 8 consecutive-n f16 values per lane -> one 16B store per tile.
// Output attn f16 in [b][384][4096] (WMMA-B layout for proj GEMM).
// ---------------------------------------------------------------------------
__global__ void __launch_bounds__(256) gemm_out(const _Float16* __restrict__ Qp,
                                                const _Float16* __restrict__ kvh,
                                                _Float16* __restrict__ attn) {
  int bh = blockIdx.x;
  int w = threadIdx.x >> 5;
  int n0 = (blockIdx.y * 8 + w) * 16;           // 256 n-tiles per (b,h)
  int b = bh >> 3, h = bh & 7;
  const _Float16* Qb = Qp + (size_t)bh * 4096 * 64;
  const _Float16* Kb = kvh + (size_t)bh * 64 * 64;
  v8f c[4] = {};
  #pragma unroll
  for (int kc = 0; kc < 64; kc += 32) {
    v16h a = load_a_frag(Qb + (size_t)n0 * 64 + kc, 64);
    #pragma unroll
    for (int e = 0; e < 4; ++e) {
      v16h bf = load_b_frag(Kb + (size_t)kc * 64 + e * 16, 64);
      c[e] = wmma_f16(a, bf, c[e]);
    }
  }
  int lane = threadIdx.x & 31;
  int col = lane & 15, rb = (lane & 16) ? 8 : 0;
  int src = lane & 16;                          // lane holding column e=48 for my rows
  #pragma unroll
  for (int e = 0; e < 3; ++e) {
    v8h pk;
    #pragma unroll
    for (int r = 0; r < 8; ++r) {
      float nr = fmaxf(__shfl(c[3][r], src, 32), 1e-6f);
      pk[r] = (_Float16)(c[e][r] / nr);
    }
    int ch = h * 48 + e * 16 + col;
    _Float16* dst = attn + ((size_t)b * 384 + ch) * 4096 + n0 + rb;
    *(v8h*)dst = pk;                            // 8 consecutive n, 16B aligned
  }
}

// ---------------------------------------------------------------------------
// Stage 5: y = BN( Wproj(384x384) @ attn(384x4096) ) per batch, fp32 out.
// Same TDM-staged structure as gemm_qkv, BN affine fused in the epilogue.
// ---------------------------------------------------------------------------
__global__ void __launch_bounds__(256) gemm_proj(const _Float16* __restrict__ W,
                                                 const _Float16* __restrict__ A,
                                                 const float* __restrict__ scale,
                                                 const float* __restrict__ shift,
                                                 float* __restrict__ out) {
  __shared__ __align__(16) _Float16 Bt[32 * 64];
  int b  = blockIdx.z;
  int w  = threadIdx.x >> 5;
  int wx = w >> 1, wy = w & 1;
  int m0 = blockIdx.x * 128 + wx * 32;
  int nb = blockIdx.y * 64;
  const _Float16* Bb = A + (size_t)b * 384 * 4096;
  u32 lds = (u32)(size_t)&Bt[0];
  v8f c00 = {}, c01 = {}, c10 = {}, c11 = {};
  for (int kc = 0; kc < 384; kc += 32) {
    __syncthreads();
    if (__builtin_amdgcn_readfirstlane(w) == 0) {
      u64 g = (u64)(size_t)(Bb + (size_t)kc * 4096 + nb);
      tdm_load_2d_f16(g, lds, /*tile_w=*/64, /*tile_h=*/32,
                      /*stride=*/4096, /*tensor_w=*/4096, /*tensor_h=*/4096);
      __builtin_amdgcn_s_wait_tensorcnt(0);
    }
    __syncthreads();
    v16h a0 = load_a_frag(W + (size_t)m0 * 384 + kc, 384);
    v16h a1 = load_a_frag(W + (size_t)(m0 + 16) * 384 + kc, 384);
    v16h b0 = load_b_frag(Bt + wy * 32, 64);
    v16h b1 = load_b_frag(Bt + wy * 32 + 16, 64);
    c00 = wmma_f16(a0, b0, c00);
    c01 = wmma_f16(a0, b1, c01);
    c10 = wmma_f16(a1, b0, c10);
    c11 = wmma_f16(a1, b1, c11);
  }
  int lane = threadIdx.x & 31;
  int col = lane & 15, rb = (lane & 16) ? 8 : 0;
  int n0 = nb + wy * 32;
  float* ob = out + (size_t)b * 384 * 4096;
  #pragma unroll
  for (int r = 0; r < 8; ++r) {
    int o0 = m0 + rb + r, o1 = m0 + 16 + rb + r;
    float s0 = scale[o0], t0 = shift[o0];
    float s1 = scale[o1], t1 = shift[o1];
    ob[(size_t)o0 * 4096 + n0 + col]      = c00[r] * s0 + t0;
    ob[(size_t)o0 * 4096 + n0 + 16 + col] = c01[r] * s0 + t0;
    ob[(size_t)o1 * 4096 + n0 + col]      = c10[r] * s1 + t1;
    ob[(size_t)o1 * 4096 + n0 + 16 + col] = c11[r] * s1 + t1;
  }
}

// ---------------------------------------------------------------------------
extern "C" void kernel_launch(void* const* d_in, const int* in_sizes, int n_in,
                              void* d_out, int out_size, void* d_ws, size_t ws_size,
                              hipStream_t stream) {
  const float* x      = (const float*)d_in[0];
  const float* w_qkv  = (const float*)d_in[1];
  const float* w_dw3  = (const float*)d_in[2];
  const float* w_dw5  = (const float*)d_in[3];
  const float* w_proj = (const float*)d_in[4];
  const float* gamma  = (const float*)d_in[5];
  const float* beta   = (const float*)d_in[6];
  const float* mean   = (const float*)d_in[7];
  const float* var    = (const float*)d_in[8];
  float* out = (float*)d_out;
  (void)in_sizes; (void)n_in; (void)out_size; (void)ws_size;

  char* ws = (char*)d_ws;
  size_t off = 0;
  auto take = [&](size_t bytes) -> char* {
    char* p = ws + off;
    off += (bytes + 255) & ~(size_t)255;
    return p;
  };
  _Float16* Xh    = (_Float16*)take((size_t)8 * 384 * 4096 * 2);    // x in f16
  _Float16* Wqh   = (_Float16*)take((size_t)1152 * 384 * 2);        // w_qkv f16
  _Float16* Wph   = (_Float16*)take((size_t)384 * 384 * 2);         // w_proj f16
  float*    scale = (float*)take(384 * 4);
  float*    shift = (float*)take(384 * 4);
  _Float16* qkvh  = (_Float16*)take((size_t)8 * 1152 * 4096 * 2);   // qkv f16 (L2-resident)
  _Float16* Qp    = (_Float16*)take((size_t)64 * 4096 * 64 * 2);    // q  [bh][n][64]
  _Float16* Kt    = (_Float16*)take((size_t)8 * 384 * 4096 * 2);    // k  [b][384][n]
  _Float16* Vp    = (_Float16*)take((size_t)64 * 4096 * 64 * 2);    // v  [bh][n][64] (+ones col)
  float*    kvacc = (float*)take((size_t)64 * 48 * 64 * 4);         // kv fp32 accum
  _Float16* kvh   = (_Float16*)take((size_t)64 * 64 * 64 * 2);      // kv f16 K-padded
  _Float16* attn  = (_Float16*)take((size_t)8 * 384 * 4096 * 2);    // attention out f16

  cvt_f32_f16<<<(12582912 + 255) / 256, 256, 0, stream>>>(x, Xh, 12582912);
  cvt_f32_f16<<<(442368 + 255) / 256, 256, 0, stream>>>(w_qkv, Wqh, 442368);
  cvt_f32_f16<<<(147456 + 255) / 256, 256, 0, stream>>>(w_proj, Wph, 147456);
  bn_prep<<<2, 256, 0, stream>>>(gamma, beta, mean, var, scale, shift);
  pad_qv<<<1024, 256, 0, stream>>>(Qp, Vp);
  zero_f32<<<768, 256, 0, stream>>>(kvacc, 196608);

  gemm_qkv<<<dim3(9, 64, 8), 256, 0, stream>>>(Wqh, Xh, qkvh);
  dwconv_route<<<147456, 256, 0, stream>>>(qkvh, w_dw3, w_dw5, Qp, Kt, Vp);
  gemm_kv<<<dim3(64, 4), 128, 0, stream>>>(Kt, Vp, kvacc);
  cvt_kv<<<1024, 256, 0, stream>>>(kvacc, kvh);
  gemm_out<<<dim3(64, 32), 256, 0, stream>>>(Qp, kvh, attn);
  gemm_proj<<<dim3(3, 64, 8), 256, 0, stream>>>(Wph, attn, scale, shift, out);
}
